// TogeGCN_90391881711981
// MI455X (gfx1250) — compile-verified, matrix operand order checked by reading, not run on hardware
//
#include <hip/hip_runtime.h>
#include <math.h>

#define DIM 128
#define EPSQ 1e-8f

typedef __attribute__((ext_vector_type(2)))  float    v2f;
typedef __attribute__((ext_vector_type(8)))  float    v8f;
typedef __attribute__((ext_vector_type(16))) _Float16 v16h;

static inline int cdiv(long a, long b) { return (int)((a + b - 1) / b); }

// ---------------- elementwise / setup kernels ----------------

__global__ void k_zero(float* __restrict__ p, int n) {
  int i = blockIdx.x * blockDim.x + threadIdx.x;
  if (i < n) p[i] = 0.0f;
}

// degree histogram: ds[src[e]] += 1, dd[dst[e]] += 1
__global__ void k_degree(const int* __restrict__ s, const int* __restrict__ d,
                         float* __restrict__ ds, float* __restrict__ dd, int e) {
  int i = blockIdx.x * blockDim.x + threadIdx.x;
  if (i < e) {
    atomicAdd(ds + s[i], 1.0f);
    atomicAdd(dd + d[i], 1.0f);
  }
}

// coef[e] = rsqrt(max(ds[s]*dd[d], 1))
__global__ void k_coef(const int* __restrict__ s, const int* __restrict__ d,
                       const float* __restrict__ ds, const float* __restrict__ dd,
                       float* __restrict__ coef, int e) {
  int i = blockIdx.x * blockDim.x + threadIdx.x;
  if (i < e) coef[i] = rsqrtf(fmaxf(ds[s[i]] * dd[d[i]], 1.0f));
}

// out[i,d] = (1 - deg[i]/(deg[i]+eps)) * emb[i,d]   (write_conv self term)
__global__ void k_selfweight(const float* __restrict__ emb, const float* __restrict__ deg,
                             float* __restrict__ out, int total) {
  int t = blockIdx.x * blockDim.x + threadIdx.x;
  if (t < total) {
    int i = t >> 7;
    float g = deg[i];
    out[t] = (1.0f - g / (g + EPSQ)) * emb[t];
  }
}

// out[wid[e], d] += scale * coef[e] * h[rid[e], d]   (one thread per (edge, channel);
// a wave covers 32 consecutive channels of one edge -> contiguous 128B load + atomics)
__global__ void k_scatter(const int* __restrict__ rid, const int* __restrict__ wid,
                          const float* __restrict__ coef, const float* __restrict__ h,
                          float* __restrict__ out, int total, float scale) {
  int t = blockIdx.x * blockDim.x + threadIdx.x;
  if (t < total) {
    int e = t >> 7;
    int d = t & 127;
    float v = scale * coef[e] * h[(size_t)rid[e] * DIM + d];
    atomicAdd(out + (size_t)wid[e] * DIM + d, v);
  }
}

// final[i,d] = aemb + 0.5*co_a1 + 0.5*(1+u_sw[i])*wr_a1   (layer-2 self terms folded)
__global__ void k_final_init(const float* __restrict__ aemb, const float* __restrict__ co_a1,
                             const float* __restrict__ wr_a1, const float* __restrict__ da,
                             float* __restrict__ fin, int total) {
  int t = blockIdx.x * blockDim.x + threadIdx.x;
  if (t < total) {
    int i = t >> 7;
    float g = da[i];
    float usw = 1.0f - g / (g + EPSQ);
    fin[t] = aemb[t] + 0.5f * co_a1[t] + 0.5f * (1.0f + usw) * wr_a1[t];
  }
}

// out[b,d] = tbl[idx[b], d]
__global__ void k_gather(const float* __restrict__ tbl, const int* __restrict__ idx,
                         float* __restrict__ out, int total) {
  int t = blockIdx.x * blockDim.x + threadIdx.x;
  if (t < total) {
    int b = t >> 7;
    int d = t & 127;
    out[t] = tbl[(size_t)idx[b] * DIM + d];
  }
}

// ---------------- WMMA scoring: predict[b] = sigmoid(dot(A[b], P[b])) ----------------
// One wave handles 16 batch rows: compute 16x16 tile of A * P^T via WMMA over K=128,
// extract the diagonal. f32 WMMA path keeps full fp32 numerics.

__global__ void k_score_wmma(const float* __restrict__ A, const float* __restrict__ P,
                             float* __restrict__ pred, int b) {
  int wave = (blockIdx.x * blockDim.x + threadIdx.x) >> 5;
  int lane = threadIdx.x & 31;
  int base = wave * 16;
  if (base >= b) return;            // wave-uniform exit (EXEC stays all-ones for WMMA)

  int row = base + (lane & 15);
  int hi  = lane >> 4;              // 0: lanes 0-15, 1: lanes 16-31
  v8f c = {};

#if __has_builtin(__builtin_amdgcn_wmma_f32_16x16x4_f32)
  // ISA 7.12.2 f32 A 16x4: lane&15 = M; v0 holds K=k0+2*hi, v1 holds K=k0+1+2*hi.
  // B 4x16 symmetric: lane&15 = N, same K striping.
  const float* ar = A + (size_t)row * DIM + 2 * hi;
  const float* pr = P + (size_t)row * DIM + 2 * hi;
#pragma unroll
  for (int k = 0; k < DIM; k += 4) {
    v2f a, bb;
    a.x = ar[k]; a.y = ar[k + 1];
    bb.x = pr[k]; bb.y = pr[k + 1];
    c = __builtin_amdgcn_wmma_f32_16x16x4_f32(false, a, false, bb, (short)0, c,
                                              false, false);
  }
#else
  // fallback: fp16 WMMA (16x16x32). ISA 7.12.2 16-bit A layout:
  // lane&15 = M; v[0..7] = K = k0 + 8*hi .. +7 ; v[8..15] = K = k0 + 16 + 8*hi .. +7
  const float* ar = A + (size_t)row * DIM + 8 * hi;
  const float* pr = P + (size_t)row * DIM + 8 * hi;
#pragma unroll
  for (int k0 = 0; k0 < DIM; k0 += 32) {
    v16h a, bb;
#pragma unroll
    for (int j = 0; j < 8; ++j) {
      a[j]      = (_Float16)ar[k0 + j];
      a[j + 8]  = (_Float16)ar[k0 + 16 + j];
      bb[j]     = (_Float16)pr[k0 + j];
      bb[j + 8] = (_Float16)pr[k0 + 16 + j];
    }
    c = __builtin_amdgcn_wmma_f32_16x16x32_f16(false, a, false, bb, (short)0, c,
                                               false, false);
  }
#endif

  // C/D layout: lanes 0-15 -> N=lane, VGPR r -> M=r ; lanes 16-31 -> N=lane-16, M=8+r.
  // Diagonal lives in lanes 0..7 (c[lane]) and lanes 24..31 (c[lane-24]).
  if (lane < 8) {
    float dv = c[lane];
    pred[base + lane] = 1.0f / (1.0f + __expf(-dv));
  } else if (lane >= 24) {
    float dv = c[lane - 24];
    pred[base + lane - 16] = 1.0f / (1.0f + __expf(-dv));
  }
}

// ---------------- driver ----------------

extern "C" void kernel_launch(void* const* d_in, const int* in_sizes, int n_in,
                              void* d_out, int out_size, void* d_ws, size_t ws_size,
                              hipStream_t stream) {
  const float* aemb = (const float*)d_in[0];
  const float* pemb = (const float*)d_in[1];
  const int* authors  = (const int*)d_in[2];
  const int* papers   = (const int*)d_in[3];
  const int* like_src = (const int*)d_in[4];
  const int* like_dst = (const int*)d_in[5];
  const int* co_src   = (const int*)d_in[6];
  const int* co_dst   = (const int*)d_in[7];

  const int NA = in_sizes[0] / DIM;
  const int NP = in_sizes[1] / DIM;
  const int B  = in_sizes[2];
  const int EL = in_sizes[4];
  const int EC = in_sizes[6];

  // workspace layout (floats)
  float* ws = (float*)d_ws;
  float* da   = ws;                 ws += NA;   // like_src out-degree
  float* dp   = ws;                 ws += NP;   // like_dst out-degree
  float* dcs  = ws;                 ws += NA;   // co_src degree
  float* dcd  = ws;                 ws += NA;   // co_dst degree
  float* ccoef = ws;                ws += EC;
  float* lcoef = ws;                ws += EL;
  float* co_a1 = ws;                ws += (size_t)NA * DIM;
  float* wr_a1 = ws;                ws += (size_t)NA * DIM;
  float* wr_p1 = ws;                ws += (size_t)NP * DIM;
  float* fin   = ws;                ws += (size_t)NA * DIM;

  float* pred = (float*)d_out;               // [B]
  float* la   = pred + B;                    // latest_author [B,128]
  float* lp   = la + (size_t)B * DIM;        // latest_paper  [B,128]

  const int T = 256;
  const int naDim = NA * DIM;
  const int elDim = EL * DIM;
  const int ecDim = EC * DIM;
  const int bDim  = B * DIM;

  // 1) degrees
  k_zero<<<cdiv(3 * NA + NP, T), T, 0, stream>>>(da, 3 * NA + NP);  // da,dp,dcs,dcd contiguous
  k_zero<<<cdiv(naDim, T), T, 0, stream>>>(co_a1, naDim);
  k_degree<<<cdiv(EL, T), T, 0, stream>>>(like_src, like_dst, da, dp, EL);
  k_degree<<<cdiv(EC, T), T, 0, stream>>>(co_src, co_dst, dcs, dcd, EC);

  // 2) edge coefficients
  k_coef<<<cdiv(EC, T), T, 0, stream>>>(co_src, co_dst, dcs, dcd, ccoef, EC);
  k_coef<<<cdiv(EL, T), T, 0, stream>>>(like_src, like_dst, da, dp, lcoef, EL);

  // 3) layer 1
  k_selfweight<<<cdiv(naDim, T), T, 0, stream>>>(aemb, da, wr_a1, naDim);
  k_selfweight<<<cdiv(NP * DIM, T), T, 0, stream>>>(pemb, dp, wr_p1, NP * DIM);
  // co_a1 = co_conv(author_emb)
  k_scatter<<<cdiv(ecDim, T), T, 0, stream>>>(co_src, co_dst, ccoef, aemb, co_a1, ecDim, 1.0f);
  // wr_a1 += scatter p[like_dst] -> like_src ; wr_p1 += scatter a[like_src] -> like_dst
  k_scatter<<<cdiv(elDim, T), T, 0, stream>>>(like_dst, like_src, lcoef, pemb, wr_a1, elDim, 1.0f);
  k_scatter<<<cdiv(elDim, T), T, 0, stream>>>(like_src, like_dst, lcoef, aemb, wr_p1, elDim, 1.0f);

  // 4) layer 2 folded directly into final_author accumulator
  k_final_init<<<cdiv(naDim, T), T, 0, stream>>>(aemb, co_a1, wr_a1, da, fin, naDim);
  k_scatter<<<cdiv(ecDim, T), T, 0, stream>>>(co_src, co_dst, ccoef, co_a1, fin, ecDim, 0.5f);
  k_scatter<<<cdiv(elDim, T), T, 0, stream>>>(like_dst, like_src, lcoef, wr_p1, fin, elDim, 0.5f);

  // 5) gathers for output
  k_gather<<<cdiv(bDim, T), T, 0, stream>>>(fin, authors, la, bDim);
  k_gather<<<cdiv(bDim, T), T, 0, stream>>>(pemb, papers, lp, bDim);

  // 6) WMMA scoring: one wave per 16 batch rows
  int waves = (B + 15) / 16;
  int blocks = cdiv((long)waves * 32, T);
  k_score_wmma<<<blocks, T, 0, stream>>>(la, lp, pred, B);
}
